// Analogy_32160715113084
// MI455X (gfx1250) — matrix operand order repaired and checked by reading, not compiled, not run on hardware
//
#include <hip/hip_runtime.h>
#include <hip/hip_bf16.h>

// ---------------------------------------------------------------------------
// RGCN layer (bdd): out = segment_sum(msg) + emb[node_ids] @ loop_weight + bias
//   Kernel 0: split loop_weight into bf16 hi/lo, transposed, into d_ws (1 MB).
//   Kernel 1: self-loop GEMM, bf16 split-precision WMMA (hi*hi+hi*lo+lo*hi).
//             Wave computes 32x64 output (2x4 tiles, 24 v_wmma / K=32 chunk);
//             B fragments are single 32B bf16 vector loads from the
//             pre-split transposed copy -> conversion VALU only for A.
//   Kernel 2: edge BDD messages, bandwidth-bound, atomic scatter-add.
// ---------------------------------------------------------------------------

#define H_DIM 512
#define NUM_BASES 128
#define SUBMAT 4

typedef __attribute__((ext_vector_type(16))) __bf16 v16bf;
typedef __attribute__((ext_vector_type(8)))  float  v8f;

// ---- A fragment (16x32 bf16 layout): lane holds row lid,
//      elems 0..15 <- K=k0+sub*16+j (contiguous, matches B).
__device__ __forceinline__ void load_a_frag(const float* __restrict__ arow,
                                            int k0, int sub,
                                            v16bf& hi, v16bf& lo) {
    #pragma unroll
    for (int g = 0; g < 2; ++g) {
        const float4 f0 = *(const float4*)(arow + k0 + g * 8 + sub * 16);
        const float4 f1 = *(const float4*)(arow + k0 + g * 8 + sub * 16 + 4);
        const float t[8] = {f0.x, f0.y, f0.z, f0.w, f1.x, f1.y, f1.z, f1.w};
        #pragma unroll
        for (int j = 0; j < 8; ++j) {
            const __bf16 h = (__bf16)t[j];
            hi[g * 8 + j] = h;
            lo[g * 8 + j] = (__bf16)(t[j] - (float)h);
        }
    }
}

// ---- B fragment from f32 (fallback path, strided column gather + split).
__device__ __forceinline__ void load_b_frag_f32(const float* __restrict__ w,
                                                int k0, int sub, int col,
                                                v16bf& hi, v16bf& lo) {
    const float* __restrict__ bcol = w + (size_t)(k0 + sub * 16) * H_DIM + col;
    #pragma unroll
    for (int i = 0; i < 16; ++i) {
        const float v = bcol[(size_t)i * H_DIM];
        const __bf16 h = (__bf16)v;
        hi[i] = h;
        lo[i] = (__bf16)(v - (float)h);
    }
}

// ---- 3-term split-precision accumulate.
__device__ __forceinline__ v8f mma3(const v16bf& ah, const v16bf& al,
                                    const v16bf& bh, const v16bf& bl, v8f c) {
    c = __builtin_amdgcn_wmma_f32_16x16x32_bf16(false, ah, false, bh,
                                                (short)0, c, false, false);
    c = __builtin_amdgcn_wmma_f32_16x16x32_bf16(false, ah, false, bl,
                                                (short)0, c, false, false);
    c = __builtin_amdgcn_wmma_f32_16x16x32_bf16(false, al, false, bh,
                                                (short)0, c, false, false);
    return c;
}

// ---- Kernel 0: loop_weight [k][col] f32 -> bT_hi/bT_lo [col][k] bf16.
//      Writes coalesced; 1 MB total, runs once per launch.
__global__ __launch_bounds__(256) void split_loop_weight(
    const float* __restrict__ w,
    __bf16*      __restrict__ bT_hi,
    __bf16*      __restrict__ bT_lo)
{
    const int idx = blockIdx.x * 256 + threadIdx.x;   // col-major over bT
    const int col = idx >> 9;
    const int k   = idx & (H_DIM - 1);
    const float v = w[(size_t)k * H_DIM + col];
    const __bf16 h = (__bf16)v;
    bT_hi[idx] = h;
    bT_lo[idx] = (__bf16)(v - (float)h);
}

// ---- Kernel 1 (fast path): wave computes 32 rows x 64 cols.
__global__ __launch_bounds__(256) void rgcn_selfloop_gemm_pre(
    const int*    __restrict__ node_ids,
    const float*  __restrict__ emb,
    const __bf16* __restrict__ bT_hi,
    const __bf16* __restrict__ bT_lo,
    const float*  __restrict__ bias,
    float*        __restrict__ out,
    int num_nodes)
{
    const int lane  = threadIdx.x & 31;
    const int wave  = threadIdx.x >> 5;
    const int blk   = blockIdx.x * 8 + wave;          // one 32x64 block / wave
    const int rowBlocks = (num_nodes + 31) >> 5;
    const int nBlocks   = rowBlocks * (H_DIM / 64);
    if (blk >= nBlocks) return;                       // wave-uniform exit

    const int m0  = (blk / (H_DIM / 64)) << 5;
    const int n0  = (blk % (H_DIM / 64)) << 6;
    const int sub = lane >> 4;
    const int lid = lane & 15;

    int mr0 = m0 + lid;        if (mr0 >= num_nodes) mr0 = num_nodes - 1;
    int mr1 = m0 + 16 + lid;   if (mr1 >= num_nodes) mr1 = num_nodes - 1;
    const float* __restrict__ arow0 = emb + (size_t)node_ids[mr0] * H_DIM;
    const float* __restrict__ arow1 = emb + (size_t)node_ids[mr1] * H_DIM;

    // Per-lane base of the pre-split transposed B columns (one per col-tile).
    const __bf16* __restrict__ bh0 = bT_hi + ((size_t)(n0 + lid) * H_DIM);
    const __bf16* __restrict__ bl0 = bT_lo + ((size_t)(n0 + lid) * H_DIM);

    v8f c0[4] = {{}, {}, {}, {}};   // rows m0..m0+15
    v8f c1[4] = {{}, {}, {}, {}};   // rows m0+16..m0+31

    for (int k0 = 0; k0 < H_DIM; k0 += 32) {
        v16bf a0h, a0l, a1h, a1l;
        load_a_frag(arow0, k0, sub, a0h, a0l);
        load_a_frag(arow1, k0, sub, a1h, a1l);

        const int koff = k0 + sub * 16;               // 32B-aligned in bT
        #pragma unroll
        for (int t = 0; t < 4; ++t) {
            const v16bf bh = *(const v16bf*)(bh0 + (size_t)t * 16 * H_DIM + koff);
            const v16bf bl = *(const v16bf*)(bl0 + (size_t)t * 16 * H_DIM + koff);
            c0[t] = mma3(a0h, a0l, bh, bl, c0[t]);
            c1[t] = mma3(a1h, a1l, bh, bl, c1[t]);
        }
    }

    // C layout: VGPR r -> row tileBase + r + sub*8, col = n0 + t*16 + lid.
    const int rb0 = m0 + sub * 8;
    const int rb1 = m0 + 16 + sub * 8;
    if (m0 + 31 < num_nodes) {                        // wave-uniform fast path
        #pragma unroll
        for (int t = 0; t < 4; ++t) {
            const int   col = n0 + t * 16 + lid;
            const float bv  = bias[col];
            #pragma unroll
            for (int r = 0; r < 8; ++r) {
                out[(size_t)(rb0 + r) * H_DIM + col] = c0[t][r] + bv;
                out[(size_t)(rb1 + r) * H_DIM + col] = c1[t][r] + bv;
            }
        }
    } else {
        #pragma unroll
        for (int t = 0; t < 4; ++t) {
            const int   col = n0 + t * 16 + lid;
            const float bv  = bias[col];
            #pragma unroll
            for (int r = 0; r < 8; ++r) {
                if (rb0 + r < num_nodes)
                    out[(size_t)(rb0 + r) * H_DIM + col] = c0[t][r] + bv;
                if (rb1 + r < num_nodes)
                    out[(size_t)(rb1 + r) * H_DIM + col] = c1[t][r] + bv;
            }
        }
    }
}

// ---- Kernel 1 (fallback, no workspace): wave computes 32x32, B split on-the-fly.
__global__ __launch_bounds__(256) void rgcn_selfloop_gemm_fb(
    const int*   __restrict__ node_ids,
    const float* __restrict__ emb,
    const float* __restrict__ loop_weight,
    const float* __restrict__ bias,
    float*       __restrict__ out,
    int num_nodes)
{
    const int lane  = threadIdx.x & 31;
    const int wave  = threadIdx.x >> 5;
    const int blk   = blockIdx.x * 8 + wave;
    const int rowBlocks = (num_nodes + 31) >> 5;
    const int nBlocks   = rowBlocks * (H_DIM / 32);
    if (blk >= nBlocks) return;

    const int m0  = (blk / (H_DIM / 32)) << 5;
    const int n0  = (blk % (H_DIM / 32)) << 5;
    const int sub = lane >> 4;
    const int lid = lane & 15;

    int mr0 = m0 + lid;        if (mr0 >= num_nodes) mr0 = num_nodes - 1;
    int mr1 = m0 + 16 + lid;   if (mr1 >= num_nodes) mr1 = num_nodes - 1;
    const float* __restrict__ arow0 = emb + (size_t)node_ids[mr0] * H_DIM;
    const float* __restrict__ arow1 = emb + (size_t)node_ids[mr1] * H_DIM;
    const int col0 = n0 + lid;
    const int col1 = n0 + 16 + lid;

    v8f c00 = {}, c01 = {}, c10 = {}, c11 = {};
    for (int k0 = 0; k0 < H_DIM; k0 += 32) {
        v16bf a0h, a0l, a1h, a1l, b0h, b0l, b1h, b1l;
        load_a_frag(arow0, k0, sub, a0h, a0l);
        load_a_frag(arow1, k0, sub, a1h, a1l);
        load_b_frag_f32(loop_weight, k0, sub, col0, b0h, b0l);
        load_b_frag_f32(loop_weight, k0, sub, col1, b1h, b1l);
        c00 = mma3(a0h, a0l, b0h, b0l, c00);
        c01 = mma3(a0h, a0l, b1h, b1l, c01);
        c10 = mma3(a1h, a1l, b0h, b0l, c10);
        c11 = mma3(a1h, a1l, b1h, b1l, c11);
    }

    const float bv0 = bias[col0];
    const float bv1 = bias[col1];
    const int   rb0 = m0 + sub * 8;
    const int   rb1 = m0 + 16 + sub * 8;
    #pragma unroll
    for (int r = 0; r < 8; ++r) {
        if (rb0 + r < num_nodes) {
            out[(size_t)(rb0 + r) * H_DIM + col0] = c00[r] + bv0;
            out[(size_t)(rb0 + r) * H_DIM + col1] = c01[r] + bv1;
        }
        if (rb1 + r < num_nodes) {
            out[(size_t)(rb1 + r) * H_DIM + col0] = c10[r] + bv0;
            out[(size_t)(rb1 + r) * H_DIM + col1] = c11[r] + bv1;
        }
    }
}

// ---- Kernel 2: one thread per (edge, output element); atomic scatter-add.
__global__ __launch_bounds__(256) void rgcn_edge_msg(
    const int*   __restrict__ node_ids,
    const int*   __restrict__ src,
    const int*   __restrict__ dst,
    const int*   __restrict__ etypes,
    const float* __restrict__ norm,
    const float* __restrict__ emb,
    const float* __restrict__ weight,
    float*       __restrict__ out,
    int num_edges)
{
    const long long gid = (long long)blockIdx.x * blockDim.x + threadIdx.x;
    const int e = (int)(gid >> 9);
    const int o = (int)(gid & (H_DIM - 1));
    if (e >= num_edges) return;

    const int s  = src[e];
    const int d  = dst[e];
    const int t  = etypes[e];
    const float nr = norm[e];

    const int b = o >> 2;
    const int j = o & 3;

    const float* __restrict__ h =
        emb + (size_t)node_ids[s] * H_DIM + b * SUBMAT;
    const float* __restrict__ w =
        weight + (((size_t)t * NUM_BASES + b) * SUBMAT) * SUBMAT + j;

    float acc = 0.f;
    #pragma unroll
    for (int i = 0; i < SUBMAT; ++i)
        acc += h[i] * w[i * SUBMAT];
    acc *= nr;

    unsafeAtomicAdd(&out[(size_t)d * H_DIM + o], acc);  // global_atomic_add_f32
}

extern "C" void kernel_launch(void* const* d_in, const int* in_sizes, int n_in,
                              void* d_out, int out_size, void* d_ws, size_t ws_size,
                              hipStream_t stream) {
    const int*   node_ids    = (const int*)d_in[0];
    const int*   src         = (const int*)d_in[1];
    const int*   dst         = (const int*)d_in[2];
    const int*   etypes      = (const int*)d_in[3];
    const float* norm        = (const float*)d_in[4];
    const float* emb         = (const float*)d_in[5];
    const float* weight      = (const float*)d_in[6];
    const float* loop_weight = (const float*)d_in[7];
    const float* bias        = (const float*)d_in[8];
    float* out = (float*)d_out;

    const int N = in_sizes[0];
    const int E = in_sizes[1];

    const size_t wsNeeded = (size_t)H_DIM * H_DIM * 2 * sizeof(__bf16); // 1 MB

    if (ws_size >= wsNeeded) {
        __bf16* bT_hi = (__bf16*)d_ws;
        __bf16* bT_lo = bT_hi + (size_t)H_DIM * H_DIM;

        // 0) Pre-split loop_weight (transposed) -> bf16 hi/lo.
        split_loop_weight<<<(H_DIM * H_DIM) / 256, 256, 0, stream>>>(
            loop_weight, bT_hi, bT_lo);

        // 1) GEMM + bias initializes every element of d_out exactly once.
        const int rowBlocks  = (N + 31) / 32;
        const int blocks     = rowBlocks * (H_DIM / 64);
        const int gemmBlocks = (blocks + 7) / 8;
        rgcn_selfloop_gemm_pre<<<gemmBlocks, 256, 0, stream>>>(
            node_ids, emb, bT_hi, bT_lo, bias, out, N);
    } else {
        const int rowBlocks  = (N + 31) / 32;
        const int blocks     = rowBlocks * (H_DIM / 32);
        const int gemmBlocks = (blocks + 7) / 8;
        rgcn_selfloop_gemm_fb<<<gemmBlocks, 256, 0, stream>>>(
            node_ids, emb, loop_weight, bias, out, N);
    }

    // 2) Edge messages accumulate on top (stream-ordered after GEMM).
    const long long work    = (long long)E * H_DIM;
    const int       eBlocks = (int)((work + 255) / 256);
    rgcn_edge_msg<<<eBlocks, 256, 0, stream>>>(
        node_ids, src, dst, etypes, norm, emb, weight, out, E);
}
